// FeedbackTransformerDecoder_2413771621118
// MI455X (gfx1250) — matrix operand order repaired and checked by reading, not compiled, not run on hardware
//
#include <hip/hip_runtime.h>
#include <hip/hip_bf16.h>
#include <math.h>

// ---------------------------------------------------------------------------
// FeedbackTransformerDecoder for MI455X (gfx1250, wave32, WMMA).
//   T=512 S=512 B=16 D=1024 H=16 DH=64 DFF=4096 L=2 MCTX=16
// Key ideas:
//  * f16 weight cache in ws (L2-resident, 192MB L2) feeding
//    v_wmma_f32_16x16x32_f16 with f32 accumulation.
//  * A-matrix (16 x K) staged once per block into LDS in WMMA-fragment
//    layout (one 32B ds read per fragment, conflict-free banking),
//    shared by all 8 waves -> 1 ds_load + 1 global_load per WMMA.
//  * mem K/V ring buffer: only the NEW state row is projected each step
//    (mathematically identical to the reference's full recompute).
//  * enc K/V projected once with WMMA into f16.
//  * attention (q-len 1) in VALU with wave32 shuffles.
// ---------------------------------------------------------------------------

typedef __attribute__((ext_vector_type(16))) _Float16 v16h;
typedef __attribute__((ext_vector_type(2)))  _Float16 v2h;
typedef __attribute__((ext_vector_type(8)))  float    v8f;

#define DEV __device__ __forceinline__

#define A_CHUNK 1024                 // K staged per LDS pass
#define A_LDS_HALFS (32 * 512)       // 32 k-blocks * (32 lanes * 16 halfs) = 32KB

// Cooperatively convert X[0:16, kc:kc+1024] (f32, row-major, ld=ldx) into
// WMMA A-fragment layout in LDS:
//   alds[kb*512 + lane*16 + e]  ==  A element (m = lane%16,
//       k = kc + kb*32 + (lane/16)*16 + e)
DEV void stage_a_chunk(const float* __restrict__ X, int ldx, int kc,
                       _Float16* __restrict__ alds) {
  for (int p = threadIdx.x; p < 8192; p += blockDim.x) {   // pairs of halfs
    int kb   = p >> 8;          // 0..31
    int r    = p & 255;
    int lane = r >> 3;          // 0..31
    int pe   = r & 7;           // pair index, e = 2*pe
    int e    = pe << 1;
    int m    = lane & 15;
    int k    = kc + kb * 32 + ((lane >> 4) << 4) + e;
    const float2 v = *(const float2*)(X + (size_t)m * ldx + k);  // k is even
    v2h hv; hv[0] = (_Float16)v.x; hv[1] = (_Float16)v.y;
    *(v2h*)(alds + kb * 512 + lane * 16 + e) = hv;
  }
}

// B fragment (32x16 f16, K x N) from row-major f16 W (N rows x K cols), B[k][n]=W[n][k].
// lane l: N = l%16 ; K = (l/16)*16 + e  -> one contiguous 32B load per lane.
DEV v16h load_b16_f16(const _Float16* __restrict__ W, int ldw, int n0, int k0) {
  int lane = threadIdx.x & 31;
  const _Float16* p = W + (size_t)(n0 + (lane & 15)) * ldw + k0 + ((lane >> 4) << 4);
  return *(const v16h*)p;
}

// Block-cooperative 16x16 tile GEMM: all waves share A (staged via LDS),
// each wave owns output tile at n0. K must be a multiple of 1024.
DEV v8f gemm_tile_blk(const float* __restrict__ X, int ldx,
                      const _Float16* __restrict__ W, int ldw,
                      int n0, int K, _Float16* __restrict__ alds) {
  v8f c = {};
  int lane = threadIdx.x & 31;
  for (int kc = 0; kc < K; kc += A_CHUNK) {
    stage_a_chunk(X, ldx, kc, alds);
    __syncthreads();
    const _Float16* ap = alds + lane * 16;
#pragma unroll 4
    for (int kb = 0; kb < 32; ++kb) {
      v16h a = *(const v16h*)(ap + kb * 512);
      v16h b = load_b16_f16(W, ldw, n0, kc + kb * 32);
      c = __builtin_amdgcn_wmma_f32_16x16x32_f16(false, a, false, b, (short)0, c, false, false);
    }
    __syncthreads();
  }
  return c;
}

// Generic 16xN GEMM: Y = act(X @ W^T + bias [+ resid]) ; f32 out.
// flags: bit0 = add resid (ld = N), bit1 = exact gelu.
__global__ void __launch_bounds__(256)
gemm16_kernel(const float* __restrict__ X, int ldx,
              const _Float16* __restrict__ W,
              const float* __restrict__ bias,
              const float* __restrict__ resid,
              float* __restrict__ Y,
              int N, int K, int flags) {
  __shared__ _Float16 alds[A_LDS_HALFS];
  int tile = blockIdx.x * (blockDim.x >> 5) + (threadIdx.x >> 5);
  int n0 = tile * 16;
  if (n0 >= N) n0 = N - 16;                    // keep barriers uniform
  v8f c = gemm_tile_blk(X, ldx, W, K, n0, K, alds);
  int lane = threadIdx.x & 31;
  int n = n0 + (lane & 15);
  int mb = (lane >> 4) << 3;
#pragma unroll
  for (int r = 0; r < 8; ++r) {
    int m = mb + r;
    float y = c[r] + bias[n];
    if (flags & 1) y += resid[(size_t)m * N + n];
    if (flags & 2) y = 0.5f * y * (1.0f + erff(y * 0.70710678118654752f));
    Y[(size_t)m * N + n] = y;
  }
}

// enc K/V projection (once): enc is (S,B,D) f32; Wkv = [cWk;cWv] f16 (2048x1024).
// Outputs f16 eK,eV laid out (B,S,D). The 8 waves of a block share one M-tile.
__global__ void __launch_bounds__(256)
enc_kv_kernel(const float* __restrict__ enc,
              const _Float16* __restrict__ Wkv,
              const float* __restrict__ bkv,
              _Float16* __restrict__ eK, _Float16* __restrict__ eV) {
  __shared__ _Float16 alds[A_LDS_HALFS];
  int gtile = blockIdx.x * 8 + (threadIdx.x >> 5);   // 512 mtiles * 128 ntiles
  int nt = gtile & 127;
  int mt = gtile >> 7;                               // uniform within block
  int b  = mt >> 5;
  int st = (mt & 31) * 16;                           // starting s index
  const float* X = enc + ((size_t)st * 16 + b) * 1024;   // row m -> s=st+m, ld = B*D
  v8f c = gemm_tile_blk(X, 16 * 1024, Wkv, 1024, nt * 16, 1024, alds);
  int lane = threadIdx.x & 31;
  int n = nt * 16 + (lane & 15);
  int mb = (lane >> 4) << 3;
#pragma unroll
  for (int r = 0; r < 8; ++r) {
    int s = st + mb + r;
    float y = c[r] + bkv[n];
    size_t o = ((size_t)b * 512 + s) * 1024;
    if (n < 1024) eK[o + n] = (_Float16)y;
    else          eV[o + n - 1024] = (_Float16)y;
  }
}

// Project new state (16x1024) through [mWk;mWv]^T into ring slot.
__global__ void __launch_bounds__(256)
state_kv_kernel(const float* __restrict__ acc,
                const _Float16* __restrict__ Wkv,
                const float* __restrict__ bkv,
                _Float16* __restrict__ mK, _Float16* __restrict__ mV, int slot) {
  __shared__ _Float16 alds[A_LDS_HALFS];
  int tile = blockIdx.x * 8 + (threadIdx.x >> 5);    // 128 tiles
  int n0 = tile * 16;
  v8f c = gemm_tile_blk(acc, 1024, Wkv, 1024, n0, 1024, alds);
  int lane = threadIdx.x & 31;
  int n = n0 + (lane & 15);
  int mb = (lane >> 4) << 3;
#pragma unroll
  for (int r = 0; r < 8; ++r) {
    int m = mb + r;                                   // batch index
    float y = c[r] + bkv[n];
    size_t o = (size_t)(m * 16 + slot) * 1024;
    if (n < 1024) mK[o + n] = (_Float16)y;
    else          mV[o + n - 1024] = (_Float16)y;
  }
}

// Memory attention: 1 query per (b,h), ring of nv=min(s+1,16) K/V rows.
// one wave per (b,h); lane covers dims e and e+32 of DH=64.
__global__ void __launch_bounds__(256)
mem_attn_kernel(const float* __restrict__ q,
                const _Float16* __restrict__ mK, const _Float16* __restrict__ mV,
                int s, float* __restrict__ out) {
  int wave = (blockIdx.x * blockDim.x + threadIdx.x) >> 5; // 0..255
  int lane = threadIdx.x & 31;
  int b = wave >> 4, h = wave & 15;
  int nv = (s + 1 < 16) ? s + 1 : 16;
  const float* qp = q + (size_t)b * 1024 + h * 64;
  float q0 = qp[lane], q1 = qp[lane + 32];
  float sc[16];
#pragma unroll
  for (int j = 0; j < 16; ++j) {
    if (j < nv) {
      int slot = (s - j) & 15;
      const _Float16* kp = mK + (size_t)(b * 16 + slot) * 1024 + h * 64;
      float p = q0 * (float)kp[lane] + q1 * (float)kp[lane + 32];
#pragma unroll
      for (int d = 16; d >= 1; d >>= 1) p += __shfl_xor(p, d, 32);
      sc[j] = p * 0.125f;                              // 1/sqrt(64)
    } else {
      sc[j] = -3.0e38f;
    }
  }
  float mx = -3.0e38f;
#pragma unroll
  for (int j = 0; j < 16; ++j) mx = fmaxf(mx, sc[j]);
  float sum = 0.0f;
#pragma unroll
  for (int j = 0; j < 16; ++j) { float e = __expf(sc[j] - mx); sc[j] = e; sum += e; }
  float inv = 1.0f / sum;
  float o0 = 0.0f, o1 = 0.0f;
#pragma unroll
  for (int j = 0; j < 16; ++j) {
    if (j < nv) {
      int slot = (s - j) & 15;
      const _Float16* vp = mV + (size_t)(b * 16 + slot) * 1024 + h * 64;
      o0 += sc[j] * (float)vp[lane];
      o1 += sc[j] * (float)vp[lane + 32];
    }
  }
  float* op = out + (size_t)b * 1024 + h * 64;
  op[lane] = o0 * inv;
  op[lane + 32] = o1 * inv;
}

// Cross attention over S=512 encoder keys; one block (256 thr) per (b,h).
__global__ void __launch_bounds__(256)
cross_attn_kernel(const float* __restrict__ q,
                  const _Float16* __restrict__ eK, const _Float16* __restrict__ eV,
                  float* __restrict__ out) {
  __shared__ float qs[64];
  __shared__ float sc[512];
  __shared__ float red[16];
  __shared__ float part[4][64];
  int b = blockIdx.x >> 4, h = blockIdx.x & 15;
  int t = threadIdx.x;
  if (t < 64) qs[t] = q[(size_t)b * 1024 + h * 64 + t];
  __syncthreads();
  const _Float16* kb = eK + (size_t)b * 512 * 1024 + h * 64;
#pragma unroll
  for (int jj = 0; jj < 2; ++jj) {
    int j = t + jj * 256;
    const _Float16* kp = kb + (size_t)j * 1024;
    float p = 0.0f;
#pragma unroll
    for (int e = 0; e < 64; ++e) p += qs[e] * (float)kp[e];
    sc[j] = p * 0.125f;
  }
  __syncthreads();
  // block max
  float m = fmaxf(sc[t], sc[t + 256]);
#pragma unroll
  for (int d = 16; d >= 1; d >>= 1) m = fmaxf(m, __shfl_xor(m, d, 32));
  if ((t & 31) == 0) red[t >> 5] = m;
  __syncthreads();
  float mx = red[0];
#pragma unroll
  for (int w = 1; w < 8; ++w) mx = fmaxf(mx, red[w]);
  float e0 = __expf(sc[t] - mx), e1 = __expf(sc[t + 256] - mx);
  float ls = e0 + e1;
#pragma unroll
  for (int d = 16; d >= 1; d >>= 1) ls += __shfl_xor(ls, d, 32);
  if ((t & 31) == 0) red[8 + (t >> 5)] = ls;
  __syncthreads();
  sc[t] = e0; sc[t + 256] = e1;
  float tot = 0.0f;
#pragma unroll
  for (int w = 0; w < 8; ++w) tot += red[8 + w];
  float inv = 1.0f / tot;
  __syncthreads();
  // weighted V: thread t handles dim e = t%64 over j-chunk (t/64)*128..+128
  int e = t & 63, ch = t >> 6;
  const _Float16* vb = eV + (size_t)b * 512 * 1024 + h * 64 + e;
  float o = 0.0f;
  for (int j = ch * 128; j < ch * 128 + 128; ++j)
    o += sc[j] * (float)vb[(size_t)j * 1024];
  part[ch][e] = o;
  __syncthreads();
  if (t < 64)
    out[(size_t)b * 1024 + h * 64 + t] =
        (part[0][t] + part[1][t] + part[2][t] + part[3][t]) * inv;
}

// LayerNorm over 16 rows of 1024 (one wave per row). Optionally accumulates
// acc = (l==0 ? 0 : acc) + mw[l]*y, and optionally writes ys row to d_out.
__global__ void __launch_bounds__(512)
ln16_kernel(const float* __restrict__ Xs,
            const float* __restrict__ g, const float* __restrict__ bb,
            float* __restrict__ emb,
            float* __restrict__ acc, const float* __restrict__ mw, int l,
            float* __restrict__ ysrow) {
  int row = threadIdx.x >> 5;
  int lane = threadIdx.x & 31;
  const float* x = Xs + (size_t)row * 1024;
  float s = 0.0f, s2 = 0.0f;
#pragma unroll
  for (int i = 0; i < 32; ++i) { float v = x[lane + i * 32]; s += v; s2 += v * v; }
#pragma unroll
  for (int d = 16; d >= 1; d >>= 1) { s += __shfl_xor(s, d, 32); s2 += __shfl_xor(s2, d, 32); }
  float mean = s * (1.0f / 1024.0f);
  float var  = s2 * (1.0f / 1024.0f) - mean * mean;
  float inv  = rsqrtf(var + 1e-5f);
  float w = (acc != nullptr) ? mw[l] : 0.0f;
#pragma unroll
  for (int i = 0; i < 32; ++i) {
    int d = lane + i * 32;
    float y = (x[d] - mean) * inv * g[d] + bb[d];
    emb[(size_t)row * 1024 + d] = y;
    if (acc) {
      float a = (l == 0) ? 0.0f : acc[(size_t)row * 1024 + d];
      acc[(size_t)row * 1024 + d] = a + w * y;
    }
    if (ysrow) ysrow[(size_t)row * 1024 + d] = y;
  }
}

// Final LN in-place over all T*B rows of d_out.
__global__ void __launch_bounds__(512)
final_ln_kernel(float* __restrict__ Y, const float* __restrict__ g,
                const float* __restrict__ bb, int rows) {
  int row = blockIdx.x * (blockDim.x >> 5) + (threadIdx.x >> 5);
  if (row >= rows) return;
  int lane = threadIdx.x & 31;
  float* x = Y + (size_t)row * 1024;
  float s = 0.0f, s2 = 0.0f;
#pragma unroll
  for (int i = 0; i < 32; ++i) { float v = x[lane + i * 32]; s += v; s2 += v * v; }
#pragma unroll
  for (int d = 16; d >= 1; d >>= 1) { s += __shfl_xor(s, d, 32); s2 += __shfl_xor(s2, d, 32); }
  float mean = s * (1.0f / 1024.0f);
  float var  = s2 * (1.0f / 1024.0f) - mean * mean;
  float inv  = rsqrtf(var + 1e-5f);
#pragma unroll
  for (int i = 0; i < 32; ++i) {
    int d = lane + i * 32;
    x[d] = (x[d] - mean) * inv * g[d] + bb[d];
  }
}

__global__ void cvt_kernel(const float* __restrict__ s, _Float16* __restrict__ d, size_t n) {
  size_t i = (size_t)blockIdx.x * blockDim.x + threadIdx.x;
  size_t stride = (size_t)gridDim.x * blockDim.x;
  for (; i < n; i += stride) d[i] = (_Float16)s[i];
}

// Ring slot 0 = K/V of the all-zero initial state: k = mbk, v = mbv.
__global__ void init_ring_kernel(const float* __restrict__ mbqkv,
                                 _Float16* __restrict__ mK, _Float16* __restrict__ mV) {
  int i = blockIdx.x * blockDim.x + threadIdx.x;   // 0..16*1024-1
  int b = i >> 10, n = i & 1023;
  mK[(size_t)(b * 16) * 1024 + n] = (_Float16)mbqkv[1024 + n];
  mV[(size_t)(b * 16) * 1024 + n] = (_Float16)mbqkv[2048 + n];
}

extern "C" void kernel_launch(void* const* d_in, const int* in_sizes, int n_in,
                              void* d_out, int out_size, void* d_ws, size_t ws_size,
                              hipStream_t stream) {
  const float* tgt        = (const float*)d_in[0];
  const float* enc        = (const float*)d_in[1];
  const float* mem_Wqkv   = (const float*)d_in[6];
  const float* mem_bqkv   = (const float*)d_in[7];
  const float* mem_Wo     = (const float*)d_in[8];
  const float* mem_bo     = (const float*)d_in[9];
  const float* mem_ng     = (const float*)d_in[10];
  const float* mem_nb     = (const float*)d_in[11];
  const float* cross_Wqkv = (const float*)d_in[12];
  const float* cross_bqkv = (const float*)d_in[13];
  const float* cross_Wo   = (const float*)d_in[14];
  const float* cross_bo   = (const float*)d_in[15];
  const float* cross_ng   = (const float*)d_in[16];
  const float* cross_nb   = (const float*)d_in[17];
  const float* W1         = (const float*)d_in[18];
  const float* b1         = (const float*)d_in[19];
  const float* W2         = (const float*)d_in[20];
  const float* b2         = (const float*)d_in[21];
  const float* ffn_ng     = (const float*)d_in[22];
  const float* ffn_nb     = (const float*)d_in[23];
  const float* mw         = (const float*)d_in[24];
  const float* fin_g      = (const float*)d_in[25];
  const float* fin_b      = (const float*)d_in[26];
  float* out = (float*)d_out;

  char* ws = (char*)d_ws;
  // f16 weight cache (24M halfs = 48MB), then enc K/V, ring, activations. ~86MB total.
  _Float16* w16  = (_Float16*)(ws);
  _Float16* eK   = (_Float16*)(ws + 50331648);
  _Float16* eV   = (_Float16*)(ws + 67108864);
  _Float16* mK   = (_Float16*)(ws + 83886080);
  _Float16* mV   = (_Float16*)(ws + 84410368);
  float* qbuf  = (float*)(ws + 84934656);
  float* aobuf = (float*)(ws + 85000192);
  float* sumb  = (float*)(ws + 85065728);
  float* emb   = (float*)(ws + 85131264);
  float* accb  = (float*)(ws + 85196800);
  float* hbuf  = (float*)(ws + 85262336);

  _Float16* w_mQ  = w16;                       // mem_Wqkv rows 0..1023
  _Float16* w_mKV = w16 + 1048576;             // rows 1024..3071 (Wk;Wv)
  _Float16* w_cQ  = w16 + 3145728;
  _Float16* w_cKV = w16 + 3145728 + 1048576;
  _Float16* w_mWo = w16 + 6291456;
  _Float16* w_cWo = w16 + 7340032;
  _Float16* w_W1  = w16 + 8388608;             // (L,4096,1024)
  _Float16* w_W2  = w16 + 16777216;            // (L,1024,4096)

  // One-time weight conversion + enc K/V + ring init
  cvt_kernel<<<1024, 256, 0, stream>>>(mem_Wqkv,   w_mQ,  (size_t)3145728);
  cvt_kernel<<<1024, 256, 0, stream>>>(cross_Wqkv, w_cQ,  (size_t)3145728);
  cvt_kernel<<<512,  256, 0, stream>>>(mem_Wo,     w_mWo, (size_t)1048576);
  cvt_kernel<<<512,  256, 0, stream>>>(cross_Wo,   w_cWo, (size_t)1048576);
  cvt_kernel<<<2048, 256, 0, stream>>>(W1,         w_W1,  (size_t)8388608);
  cvt_kernel<<<2048, 256, 0, stream>>>(W2,         w_W2,  (size_t)8388608);
  init_ring_kernel<<<64, 256, 0, stream>>>(mem_bqkv, mK, mV);
  enc_kv_kernel<<<8192, 256, 0, stream>>>(enc, w_cKV, cross_bqkv + 1024, eK, eV);

  for (int s = 0; s < 512; ++s) {
    const float* embStep = tgt + (size_t)s * 16 * 1024;
    for (int l = 0; l < 2; ++l) {
      const float* e0 = (l == 0) ? embStep : emb;
      // self (memory) attention block
      gemm16_kernel<<<8, 256, 0, stream>>>(e0, 1024, w_mQ, mem_bqkv, nullptr, qbuf, 1024, 1024, 0);
      mem_attn_kernel<<<32, 256, 0, stream>>>(qbuf, mK, mV, s, aobuf);
      gemm16_kernel<<<8, 256, 0, stream>>>(aobuf, 1024, w_mWo, mem_bo, e0, sumb, 1024, 1024, 1);
      ln16_kernel<<<1, 512, 0, stream>>>(sumb, mem_ng, mem_nb, emb, nullptr, nullptr, 0, nullptr);
      // cross attention block
      gemm16_kernel<<<8, 256, 0, stream>>>(emb, 1024, w_cQ, cross_bqkv, nullptr, qbuf, 1024, 1024, 0);
      cross_attn_kernel<<<256, 256, 0, stream>>>(qbuf, eK, eV, aobuf);
      gemm16_kernel<<<8, 256, 0, stream>>>(aobuf, 1024, w_cWo, cross_bo, emb, sumb, 1024, 1024, 1);
      ln16_kernel<<<1, 512, 0, stream>>>(sumb, cross_ng, cross_nb, emb, nullptr, nullptr, 0, nullptr);
      // FFN
      gemm16_kernel<<<32, 256, 0, stream>>>(emb, 1024, w_W1 + (size_t)l * 4194304,
                                            b1 + l * 4096, nullptr, hbuf, 4096, 1024, 2);
      gemm16_kernel<<<8, 256, 0, stream>>>(hbuf, 4096, w_W2 + (size_t)l * 4194304,
                                           b2 + l * 1024, emb, sumb, 1024, 4096, 1);
      ln16_kernel<<<1, 512, 0, stream>>>(sumb, ffn_ng + l * 1024, ffn_nb + l * 1024, emb,
                                         accb, mw, l,
                                         (l == 1) ? out + (size_t)s * 16 * 1024 : nullptr);
    }
    // project new state into ring slot for next step
    state_kv_kernel<<<16, 256, 0, stream>>>(accb, w_mKV, mem_bqkv + 1024, mK, mV, (s + 1) & 15);
  }
  final_ln_kernel<<<512, 512, 0, stream>>>(out, fin_g, fin_b, 8192);
}
